// Model_2499670966276
// MI455X (gfx1250) — matrix operand order repaired and checked by reading, not compiled, not run on hardware
//
#include <hip/hip_runtime.h>
#include <hip/hip_bf16.h>
#include <math.h>

// Model dims (compile-time; match reference)
#define BB   2
#define TT   1024
#define DD   1024
#define VV   32000
#define HH   16
#define LL   8
#define FF   4096
#define HD   64

typedef _Float16 v4h  __attribute__((ext_vector_type(4)));
typedef _Float16 v8h  __attribute__((ext_vector_type(8)));
typedef _Float16 v16h __attribute__((ext_vector_type(16)));
typedef float    v8f  __attribute__((ext_vector_type(8)));

// ---------------------------------------------------------------------------
// CDNA5 async global->LDS copy (ASYNCcnt-tracked, no VGPR round trip).
// Operand form follows VGLOBAL: dst = LDS byte address VGPR, v[addr] pair, off.
// ---------------------------------------------------------------------------
__device__ __forceinline__ void async_ld_lds_b128(void* lds_ptr, const void* gptr) {
    unsigned lds_off = (unsigned)(unsigned long long)lds_ptr;  // flat LDS addr[31:0]
    unsigned long long ga = (unsigned long long)gptr;
    asm volatile("global_load_async_to_lds_b128 %0, %1, off"
                 :: "v"(lds_off), "v"(ga) : "memory");
}
__device__ __forceinline__ void wait_async0() {
    asm volatile("s_wait_asynccnt 0" ::: "memory");
}

// Build a 16-half A fragment from two 16-byte LDS chunks (ISA K-chunk layout).
__device__ __forceinline__ v16h ld_frag2(const _Float16* p0, const _Float16* p1) {
    v8h lo = *(const v8h*)p0;
    v8h hi = *(const v8h*)p1;
    v16h r;
#pragma unroll
    for (int i = 0; i < 8; ++i) { r[i] = lo[i]; r[i + 8] = hi[i]; }
    return r;
}

__device__ __forceinline__ float half16_max(float v) {
    v = fmaxf(v, __shfl_xor(v, 1, 32));
    v = fmaxf(v, __shfl_xor(v, 2, 32));
    v = fmaxf(v, __shfl_xor(v, 4, 32));
    v = fmaxf(v, __shfl_xor(v, 8, 32));
    return v;
}
__device__ __forceinline__ float half16_sum(float v) {
    v += __shfl_xor(v, 1, 32);
    v += __shfl_xor(v, 2, 32);
    v += __shfl_xor(v, 4, 32);
    v += __shfl_xor(v, 8, 32);
    return v;
}

// ---------------------------------------------------------------------------
// Embedding + sinusoidal PE:  h[b,t,:] = emb[x[b,t],:] + pe[t,:]
// ---------------------------------------------------------------------------
__global__ __launch_bounds__(256) void tfm_embed(const int* __restrict__ x,
                                                 const float* __restrict__ emb,
                                                 const float* __restrict__ pe,
                                                 float* __restrict__ h) {
    int bt  = blockIdx.x;          // 0 .. B*T-1
    int t   = bt & (TT - 1);
    int tok = x[bt];
    int d   = threadIdx.x * 4;
    float4 e = *(const float4*)(emb + (size_t)tok * DD + d);
    float4 p = *(const float4*)(pe + (size_t)t * DD + d);
    float4 o;
    o.x = e.x + p.x; o.y = e.y + p.y; o.z = e.z + p.z; o.w = e.w + p.w;
    *(float4*)(h + (size_t)bt * DD + d) = o;
}

// ---------------------------------------------------------------------------
// f32 -> f16 streaming convert (n multiple of 1024); one pass per tensor use.
// ---------------------------------------------------------------------------
__global__ __launch_bounds__(256) void tfm_cvt_f16(const float* __restrict__ src,
                                                   _Float16* __restrict__ dst) {
    size_t i = (size_t)blockIdx.x * 1024 + threadIdx.x * 4;
    float4 v = *(const float4*)(src + i);
    v4h o;
    o[0] = (_Float16)v.x; o[1] = (_Float16)v.y;
    o[2] = (_Float16)v.z; o[3] = (_Float16)v.w;
    *(v4h*)(dst + i) = o;
}

// ---------------------------------------------------------------------------
// WMMA GEMM (f16 A, f16 W):  C[M,N] = act( A @ W + bias )
// Block tile 128x64, 8 waves, wave = 16 rows x 64 cols (4 wmma accumulators).
// Both tiles staged with GLOBAL_LOAD_ASYNC_TO_LDS_B128 (pure copy, no VALU).
// act: 0 = none, 1 = exact GELU
// ---------------------------------------------------------------------------
#define GM_TM 128
#define GM_TN 64
#define GM_TK 32

__global__ __launch_bounds__(256) void tfm_gemm_wmma_f16(const _Float16* __restrict__ A,
                                                         const _Float16* __restrict__ W,
                                                         const float* __restrict__ bias,
                                                         float* __restrict__ C,
                                                         int M, int N, int K, int act) {
    __shared__ __align__(32) _Float16 As[GM_TM][GM_TK]; // [m][k]   8 KB
    __shared__ __align__(32) _Float16 Bs[GM_TK][GM_TN]; // [k][n]   4 KB

    const int tid  = threadIdx.x;
    const int wave = tid >> 5;
    const int lane = tid & 31;
    const int hs   = lane >> 4;   // K-chunk half select
    const int rm   = lane & 15;   // M row (A) / N col (C) within tile

    const int row0 = blockIdx.y * GM_TM;
    const int col0 = blockIdx.x * GM_TN;

    v8f acc[4];
#pragma unroll
    for (int j = 0; j < 4; ++j)
#pragma unroll
        for (int i = 0; i < 8; ++i) acc[j][i] = 0.0f;

    // A tile: 128x32 halfs = 512 x 16B chunks -> 2 chunks/thread (same row)
    const int ar  = tid >> 1;             // 0..127
    const int ak  = (tid & 1) * 16;       // 0 or 16 (two b128 = 16 halfs)
    // B tile: 32x64 halfs = 256 x 16B chunks -> 1 chunk/thread
    const int bk  = tid >> 3;             // 0..31
    const int bn  = (tid & 7) * 8;        // 0..56

    for (int k0 = 0; k0 < K; k0 += GM_TK) {
        __syncthreads();
        {   // async stage A tile (two b128) and B tile (one b128)
            const _Float16* ap = A + (size_t)(row0 + ar) * K + (k0 + ak);
            async_ld_lds_b128(&As[ar][ak],     ap);
            async_ld_lds_b128(&As[ar][ak + 8], ap + 8);
            const _Float16* bp = W + (size_t)(k0 + bk) * N + (col0 + bn);
            async_ld_lds_b128(&Bs[bk][bn], bp);
            if (k0 + GM_TK < K) {
                __builtin_prefetch(ap + GM_TK, 0, 1);
                __builtin_prefetch(bp + (size_t)GM_TK * N, 0, 1);
            }
        }
        wait_async0();
        __syncthreads();

        // A fragment: lane rm = M row, half-chunks {hs*8, 16+hs*8}
        v16h afrag = ld_frag2(&As[wave * 16 + rm][hs * 8],
                              &As[wave * 16 + rm][16 + hs * 8]);
#pragma unroll
        for (int j = 0; j < 4; ++j) {
            // B fragment: lane rm = N col, K rows {hs*8.., 16+hs*8..}
            v16h bfrag;
#pragma unroll
            for (int i = 0; i < 8; ++i) {
                bfrag[i]     = Bs[hs * 8 + i][j * 16 + rm];
                bfrag[i + 8] = Bs[16 + hs * 8 + i][j * 16 + rm];
            }
            acc[j] = __builtin_amdgcn_wmma_f32_16x16x32_f16(
                false, afrag, false, bfrag, (short)0, acc[j], false, false);
        }
    }

    // epilogue: C layout row = g + 8*hs, col = rm  (per 16x16 tile)
    const int rowg = row0 + wave * 16 + hs * 8;
#pragma unroll
    for (int j = 0; j < 4; ++j) {
        int col  = col0 + j * 16 + rm;
        float bv = bias ? bias[col] : 0.0f;
#pragma unroll
        for (int g = 0; g < 8; ++g) {
            float xv = acc[j][g] + bv;
            if (act == 1) xv = 0.5f * xv * (1.0f + erff(xv * 0.70710678118654752f));
            C[(size_t)(rowg + g) * N + col] = xv;
        }
    }
}

// ---------------------------------------------------------------------------
// WMMA GEMM (f32 inputs, converted while staging) — used for the LM head,
// whose weight is read ~once per M-block and stays L2-resident in f32.
// ---------------------------------------------------------------------------
__global__ __launch_bounds__(256) void tfm_gemm_wmma_f32(const float* __restrict__ A,
                                                         const float* __restrict__ W,
                                                         const float* __restrict__ bias,
                                                         float* __restrict__ C,
                                                         int M, int N, int K, int act) {
    __shared__ __align__(32) _Float16 As[GM_TM][GM_TK];
    __shared__ __align__(32) _Float16 Bs[GM_TK][GM_TN];

    const int tid  = threadIdx.x;
    const int wave = tid >> 5;
    const int lane = tid & 31;
    const int hs   = lane >> 4;
    const int rm   = lane & 15;

    const int row0 = blockIdx.y * GM_TM;
    const int col0 = blockIdx.x * GM_TN;

    v8f acc[4];
#pragma unroll
    for (int j = 0; j < 4; ++j)
#pragma unroll
        for (int i = 0; i < 8; ++i) acc[j][i] = 0.0f;

    const int ar  = tid >> 1;
    const int ako = (tid & 1) * 16;
    const int bk  = tid >> 3;
    const int bn  = (tid & 7) * 8;

    for (int k0 = 0; k0 < K; k0 += GM_TK) {
        __syncthreads();
        {   // stage A tile (f32 -> f16)
            const float* ap = A + (size_t)(row0 + ar) * K + (k0 + ako);
            if (k0 + GM_TK < K) __builtin_prefetch(ap + GM_TK, 0, 1);
            v8h h0, h1;
#pragma unroll
            for (int i = 0; i < 8; ++i) { h0[i] = (_Float16)ap[i]; h1[i] = (_Float16)ap[8 + i]; }
            *(v8h*)&As[ar][ako]     = h0;
            *(v8h*)&As[ar][ako + 8] = h1;
        }
        {   // stage B tile (f32 -> f16), [k][n] row-major
            const float* bp = W + (size_t)(k0 + bk) * N + (col0 + bn);
            if (k0 + GM_TK < K) __builtin_prefetch(bp + (size_t)GM_TK * N, 0, 1);
            v8h hb;
#pragma unroll
            for (int i = 0; i < 8; ++i) hb[i] = (_Float16)bp[i];
            *(v8h*)&Bs[bk][bn] = hb;
        }
        __syncthreads();

        v16h afrag = ld_frag2(&As[wave * 16 + rm][hs * 8],
                              &As[wave * 16 + rm][16 + hs * 8]);
#pragma unroll
        for (int j = 0; j < 4; ++j) {
            v16h bfrag;
#pragma unroll
            for (int i = 0; i < 8; ++i) {
                bfrag[i]     = Bs[hs * 8 + i][j * 16 + rm];
                bfrag[i + 8] = Bs[16 + hs * 8 + i][j * 16 + rm];
            }
            acc[j] = __builtin_amdgcn_wmma_f32_16x16x32_f16(
                false, afrag, false, bfrag, (short)0, acc[j], false, false);
        }
    }

    const int rowg = row0 + wave * 16 + hs * 8;
#pragma unroll
    for (int j = 0; j < 4; ++j) {
        int col  = col0 + j * 16 + rm;
        float bv = bias ? bias[col] : 0.0f;
#pragma unroll
        for (int g = 0; g < 8; ++g) {
            float xv = acc[j][g] + bv;
            if (act == 1) xv = 0.5f * xv * (1.0f + erff(xv * 0.70710678118654752f));
            C[(size_t)(rowg + g) * N + col] = xv;
        }
    }
}

// ---------------------------------------------------------------------------
// Flash attention (causal), hd = 64, per-head.
// Block = (b*H + h, 128 q-rows). 8 waves, each a 16-row q tile.
// K/V staged per 32-key tile; streaming softmax; ctx via P@V wmma.
// ---------------------------------------------------------------------------
__global__ __launch_bounds__(256) void tfm_attn_flash(const float* __restrict__ Q,
                                                      const float* __restrict__ Kg,
                                                      const float* __restrict__ Vg,
                                                      float* __restrict__ O) {
    __shared__ __align__(32) _Float16 k_lds[HD][32];      // [hd][key]   4 KB
    __shared__ __align__(32) _Float16 v_lds[32][HD];      // [key][hd]   4 KB
    __shared__ __align__(32) _Float16 p_lds[8][16][32];   // per-wave P  8 KB

    const int tid  = threadIdx.x;
    const int wave = tid >> 5;
    const int lane = tid & 31;
    const int hs   = lane >> 4;
    const int rm   = lane & 15;

    const int bh = blockIdx.x;      // b*H + head
    const int b  = bh >> 4;         // H == 16
    const int hh = bh & 15;
    const int q0 = blockIdx.y * 128 + wave * 16;

    // Q fragments (loop invariant): A[16 x 64] as two K-chunks of 32
    const float* qrow = Q + (size_t)(b * TT + q0 + rm) * DD + hh * HD;
    v16h qa[2];
#pragma unroll
    for (int c = 0; c < 2; ++c) {
        const float* qp = qrow + c * 32 + hs * 8;
        v16h tq;
#pragma unroll
        for (int i = 0; i < 8; ++i) {
            tq[i]     = (_Float16)qp[i];
            tq[i + 8] = (_Float16)qp[16 + i];
        }
        qa[c] = tq;
    }

    float mrow[8], lrow[8];
    v8f ctx[4];
#pragma unroll
    for (int g = 0; g < 8; ++g) { mrow[g] = -3.0e38f; lrow[g] = 0.0f; }
#pragma unroll
    for (int j = 0; j < 4; ++j)
#pragma unroll
        for (int i = 0; i < 8; ++i) ctx[j][i] = 0.0f;

    const int ktEndBlk = blockIdx.y * 4 + 3;      // block-uniform loop bound
    const int ktEndW   = (q0 + 15) >> 5;          // wave causal bound

    const int ld_key = tid >> 3;                  // 0..31
    const int ld_h0  = (tid & 7) * 8;             // 0..56

    for (int kt = 0; kt <= ktEndBlk; ++kt) {
        __syncthreads();
        {   // stage K (transposed [hd][key]) and V ([key][hd]) tiles, f32->f16
            const size_t rbase = (size_t)(b * TT + kt * 32 + ld_key) * DD + hh * HD + ld_h0;
            const float* kp = Kg + rbase;
            const float* vp = Vg + rbase;
#pragma unroll
            for (int i = 0; i < 8; ++i) k_lds[ld_h0 + i][ld_key] = (_Float16)kp[i];
            v8h vv;
#pragma unroll
            for (int i = 0; i < 8; ++i) vv[i] = (_Float16)vp[i];
            *(v8h*)&v_lds[ld_key][ld_h0] = vv;
        }
        __syncthreads();

        if (kt > ktEndW) continue;   // no barrier below -> divergence is safe

        // S = Q @ K^T : two 16x16 score tiles over 32 keys, matmul-K = hd = 64
        v8f s0, s1;
#pragma unroll
        for (int i = 0; i < 8; ++i) { s0[i] = 0.0f; s1[i] = 0.0f; }
#pragma unroll
        for (int c = 0; c < 2; ++c) {
            v16h kb0, kb1;
#pragma unroll
            for (int i = 0; i < 8; ++i) {
                kb0[i]     = k_lds[c * 32 + hs * 8 + i][rm];
                kb0[i + 8] = k_lds[c * 32 + 16 + hs * 8 + i][rm];
                kb1[i]     = k_lds[c * 32 + hs * 8 + i][16 + rm];
                kb1[i + 8] = k_lds[c * 32 + 16 + hs * 8 + i][16 + rm];
            }
            s0 = __builtin_amdgcn_wmma_f32_16x16x32_f16(false, qa[c], false, kb0,
                                                        (short)0, s0, false, false);
            s1 = __builtin_amdgcn_wmma_f32_16x16x32_f16(false, qa[c], false, kb1,
                                                        (short)0, s1, false, false);
        }

        // streaming softmax per row (row = g + 8*hs within tile)
        const int colb = kt * 32;
#pragma unroll
        for (int g = 0; g < 8; ++g) {
            int row = q0 + g + hs * 8;
            float v0 = s0[g] * 0.125f;   // 1/sqrt(64)
            float v1 = s1[g] * 0.125f;
            if (colb + rm      > row) v0 = -1.0e30f;
            if (colb + 16 + rm > row) v1 = -1.0e30f;
            float mx = half16_max(fmaxf(v0, v1));
            float mn = fmaxf(mrow[g], mx);
            float sc = __expf(mrow[g] - mn);
            float p0 = __expf(v0 - mn);
            float p1 = __expf(v1 - mn);
            lrow[g] = lrow[g] * sc + half16_sum(p0 + p1);
            mrow[g] = mn;
#pragma unroll
            for (int j = 0; j < 4; ++j) ctx[j][g] *= sc;
            p_lds[wave][g + hs * 8][rm]      = (_Float16)p0;
            p_lds[wave][g + hs * 8][16 + rm] = (_Float16)p1;
        }
        __builtin_amdgcn_wave_barrier();   // keep LDS write->read ordering

        // ctx += P[16x32] @ V[32x64]
        v16h pa = ld_frag2(&p_lds[wave][rm][hs * 8],
                           &p_lds[wave][rm][16 + hs * 8]);
#pragma unroll
        for (int j = 0; j < 4; ++j) {
            v16h vb;
#pragma unroll
            for (int i = 0; i < 8; ++i) {
                vb[i]     = v_lds[hs * 8 + i][j * 16 + rm];
                vb[i + 8] = v_lds[16 + hs * 8 + i][j * 16 + rm];
            }
            ctx[j] = __builtin_amdgcn_wmma_f32_16x16x32_f16(false, pa, false, vb,
                                                            (short)0, ctx[j], false, false);
        }
    }

    // normalize and write ctx -> O[b, q, hh*64 + :]
#pragma unroll
    for (int j = 0; j < 4; ++j) {
        int col = hh * HD + j * 16 + rm;
#pragma unroll
        for (int g = 0; g < 8; ++g) {
            int row = q0 + g + hs * 8;
            O[(size_t)(b * TT + row) * DD + col] = ctx[j][g] / lrow[g];
        }
    }
}

// ---------------------------------------------------------------------------
// Residual + LayerNorm:  Out[row,:] = LN(X[row,:] + Rsd[row,:]) * g + b
// One block per row of D=1024; Rsd may be nullptr. Out may alias X.
// ---------------------------------------------------------------------------
__global__ __launch_bounds__(256) void tfm_ln(const float* __restrict__ X,
                                              const float* __restrict__ Rsd,
                                              const float* __restrict__ gam,
                                              const float* __restrict__ bet,
                                              float* __restrict__ Out) {
    __shared__ float s_sum[256];
    __shared__ float s_sq[256];
    const int row = blockIdx.x;
    const int t   = threadIdx.x;
    const size_t base = (size_t)row * DD;
    const int d0 = t * 4;

    float v[4];
    float4 xv = *(const float4*)(X + base + d0);
    v[0] = xv.x; v[1] = xv.y; v[2] = xv.z; v[3] = xv.w;
    if (Rsd) {
        float4 rv = *(const float4*)(Rsd + base + d0);
        v[0] += rv.x; v[1] += rv.y; v[2] += rv.z; v[3] += rv.w;
    }
    float su = v[0] + v[1] + v[2] + v[3];
    float sq = v[0] * v[0] + v[1] * v[1] + v[2] * v[2] + v[3] * v[3];
    s_sum[t] = su; s_sq[t] = sq;
    __syncthreads();
    for (int off = 128; off > 0; off >>= 1) {
        if (t < off) { s_sum[t] += s_sum[t + off]; s_sq[t] += s_sq[t + off]; }
        __syncthreads();
    }
    const float mean = s_sum[0] * (1.0f / DD);
    const float var  = s_sq[0] * (1.0f / DD) - mean * mean;
    const float inv  = rsqrtf(var + 1e-5f);
#pragma unroll
    for (int i = 0; i < 4; ++i) {
        int d = d0 + i;
        Out[base + d] = (v[i] - mean) * inv * gam[d] + bet[d];
    }
}

// ---------------------------------------------------------------------------
// Orchestration
// ---------------------------------------------------------------------------
extern "C" void kernel_launch(void* const* d_in, const int* in_sizes, int n_in,
                              void* d_out, int out_size, void* d_ws, size_t ws_size,
                              hipStream_t stream) {
    (void)in_sizes; (void)n_in; (void)out_size; (void)ws_size;

    const int*   x     = (const int*)d_in[0];
    const float* emb   = (const float*)d_in[2];
    const float* pe    = (const float*)d_in[3];
    const float* Wq    = (const float*)d_in[4];
    const float* Wk    = (const float*)d_in[5];
    const float* Wv    = (const float*)d_in[6];
    const float* Wo    = (const float*)d_in[7];
    const float* bo    = (const float*)d_in[8];
    const float* ln1g  = (const float*)d_in[9];
    const float* ln1b  = (const float*)d_in[10];
    const float* W1    = (const float*)d_in[11];
    const float* b1    = (const float*)d_in[12];
    const float* W2    = (const float*)d_in[13];
    const float* b2    = (const float*)d_in[14];
    const float* ln2g  = (const float*)d_in[15];
    const float* ln2b  = (const float*)d_in[16];
    const float* lnfg  = (const float*)d_in[17];
    const float* lnfb  = (const float*)d_in[18];
    const float* Wout  = (const float*)d_in[19];
    const float* bout  = (const float*)d_in[20];
    float* out = (float*)d_out;

    const size_t BTD = (size_t)BB * TT * DD;     // 2M elems
    const size_t BTF = (size_t)BB * TT * FF;     // 8M elems
    float* ws  = (float*)d_ws;
    float* h   = ws;              // [B,T,D]  f32
    float* q   = h   + BTD;
    float* k   = q   + BTD;
    float* v   = k   + BTD;
    float* ctx = v   + BTD;
    float* tmp = ctx + BTD;
    float* ff  = tmp + BTD;       // [B,T,4D] f32
    _Float16* aX = (_Float16*)(ff + BTF);              // activation f16 [B,T,D]
    _Float16* aF = aX + BTD;                           // activation f16 [B,T,4D]
    _Float16* wF = aF + BTF;                           // weight f16 (<= D*FF)

    const int M = BB * TT;                         // 2048
    const dim3 blk(256);
    const dim3 gD (DD / GM_TN,  M / GM_TM);        // N=1024 GEMMs: (16,16)
    const dim3 gF (FF / GM_TN,  M / GM_TM);        // N=4096 GEMM:  (64,16)
    const dim3 gV (VV / GM_TN,  M / GM_TM);        // N=32000 GEMM: (500,16)
    const dim3 gA (BB * HH, TT / 128);             // attention blocks
    const int cDD  = (int)((size_t)DD * DD / 1024);  // cvt grids
    const int cDF  = (int)((size_t)DD * FF / 1024);
    const int cBTD = (int)(BTD / 1024);
    const int cBTF = (int)(BTF / 1024);

    tfm_embed<<<BB * TT, blk, 0, stream>>>(x, emb, pe, h);

    for (int l = 0; l < LL; ++l) {
        const float* wq = Wq + (size_t)l * DD * DD;
        const float* wk = Wk + (size_t)l * DD * DD;
        const float* wv = Wv + (size_t)l * DD * DD;
        const float* wo = Wo + (size_t)l * DD * DD;
        const float* w1 = W1 + (size_t)l * DD * FF;
        const float* w2 = W2 + (size_t)l * FF * DD;

        // QKV projections: convert h once, each weight once, then f16 WMMA GEMM
        tfm_cvt_f16<<<cBTD, blk, 0, stream>>>(h, aX);
        tfm_cvt_f16<<<cDD, blk, 0, stream>>>(wq, wF);
        tfm_gemm_wmma_f16<<<gD, blk, 0, stream>>>(aX, wF, nullptr, q, M, DD, DD, 0);
        tfm_cvt_f16<<<cDD, blk, 0, stream>>>(wk, wF);
        tfm_gemm_wmma_f16<<<gD, blk, 0, stream>>>(aX, wF, nullptr, k, M, DD, DD, 0);
        tfm_cvt_f16<<<cDD, blk, 0, stream>>>(wv, wF);
        tfm_gemm_wmma_f16<<<gD, blk, 0, stream>>>(aX, wF, nullptr, v, M, DD, DD, 0);

        tfm_attn_flash<<<gA, blk, 0, stream>>>(q, k, v, ctx);

        tfm_cvt_f16<<<cBTD, blk, 0, stream>>>(ctx, aX);
        tfm_cvt_f16<<<cDD, blk, 0, stream>>>(wo, wF);
        tfm_gemm_wmma_f16<<<gD, blk, 0, stream>>>(aX, wF, bo + (size_t)l * DD, tmp, M, DD, DD, 0);
        tfm_ln<<<M, blk, 0, stream>>>(h, tmp, ln1g + (size_t)l * DD, ln1b + (size_t)l * DD, h);

        // FFN
        tfm_cvt_f16<<<cBTD, blk, 0, stream>>>(h, aX);
        tfm_cvt_f16<<<cDF, blk, 0, stream>>>(w1, wF);
        tfm_gemm_wmma_f16<<<gF, blk, 0, stream>>>(aX, wF, b1 + (size_t)l * FF, ff, M, FF, DD, 1);
        tfm_cvt_f16<<<cBTF, blk, 0, stream>>>(ff, aF);
        tfm_cvt_f16<<<cDF, blk, 0, stream>>>(w2, wF);
        tfm_gemm_wmma_f16<<<gD, blk, 0, stream>>>(aF, wF, b2 + (size_t)l * DD, tmp, M, DD, FF, 0);
        tfm_ln<<<M, blk, 0, stream>>>(h, tmp, ln2g + (size_t)l * DD, ln2b + (size_t)l * DD, h);
    }

    tfm_ln<<<M, blk, 0, stream>>>(h, nullptr, lnfg, lnfb, tmp);
    tfm_gemm_wmma_f32<<<gV, blk, 0, stream>>>(tmp, Wout, bout, out, M, VV, DD, 0);
}
